// Kernel_4272043718162930052_53472342835836
// MI455X (gfx1250) — compile-verified, matrix operand order checked by reading, not run on hardware
//
#include <hip/hip_runtime.h>
#include <hip/hip_bf16.h>

typedef __attribute__((ext_vector_type(16))) _Float16 v16h;
typedef __attribute__((ext_vector_type(8)))  _Float16 v8h;
typedef __attribute__((ext_vector_type(8)))  float    v8f;

// Problem constants (n,c,h,w) = (8,256,56,56), K=7
#define NB   8
#define CC   256
#define HH   56
#define WW   56
#define HW   3136          // 56*56
#define KT   7             // taps
#define NK   1792          // c*7
#define SCALE1 0.0178571428571f   // 1/sqrt(hw) = 1/56
#define SCALE2 0.0236227795091f   // 1/sqrt(c*K) = 1/sqrt(1792)

// ---------------- prep kernels (pure bandwidth) ----------------

__global__ void prep_xh(const float* __restrict__ x, _Float16* __restrict__ xh) {
    long long i = (long long)blockIdx.x * blockDim.x + threadIdx.x;
    if (i < (long long)NB * CC * HW) xh[i] = (_Float16)x[i];
}

// t5[n, (j,k), p] = t1[n,j,p] - t1[n,(j-1)%c, p + (k-3)]   (0 outside row)
__global__ void prep_t5(const float* __restrict__ x, const float* __restrict__ p1w,
                        _Float16* __restrict__ t5) {
    long long idx = (long long)blockIdx.x * blockDim.x + threadIdx.x;
    if (idx >= (long long)NB * NK * HW) return;
    int p   = (int)(idx % HW);
    long long r = idx / HW;
    int k   = (int)(r % KT); r /= KT;
    int j   = (int)(r % CC);
    int n   = (int)(r / CC);
    int wcl = p % WW;

    float t1 = p1w[(size_t)j * HW + p] * x[((size_t)n * CC + j) * HW + p];
    float sh = 0.0f;
    int wc = wcl + k - 3;
    if (wc >= 0 && wc < WW) {
        int jm = (j + CC - 1) & (CC - 1);
        int p2 = p + (k - 3);
        sh = p1w[(size_t)jm * HW + p2] * x[((size_t)n * CC + jm) * HW + p2];
    }
    t5[idx] = (_Float16)(t1 - sh);
}

// t6T[n, p, (j,k)] = x[n,j,p] + t1[n,(j-1)%c, hh*56 + wcol]
__global__ void prep_t6t(const float* __restrict__ x, const float* __restrict__ p1w,
                         _Float16* __restrict__ t6t) {
    long long idx = (long long)blockIdx.x * blockDim.x + threadIdx.x;
    if (idx >= (long long)NB * HW * NK) return;
    int jk = (int)(idx % NK);
    long long r = idx / NK;
    int p  = (int)(r % HW);
    int n  = (int)(r / HW);
    int j  = jk / KT;
    int k  = jk % KT;
    int h_ = p / WW, w_ = p % WW;

    float v = x[((size_t)n * CC + j) * HW + p];
    int w1 = (w_ + 1) % WW;
    int wc = w1 + k - 3;
    if (wc >= 0 && wc < WW) {
        int hh = (h_ + HH - 1) % HH;
        int jm = (j + CC - 1) & (CC - 1);
        int p2 = hh * WW + wc;
        v += p1w[(size_t)jm * HW + p2] * x[((size_t)n * CC + jm) * HW + p2];
    }
    t6t[idx] = (_Float16)v;
}

// ---------------- WMMA GEMMs with async-LDS staged B ----------------

// Fragment from a row-major K-contiguous GLOBAL matrix.
// lanes 0-15 : row r,  K = {k0..k0+7, k0+16..k0+23}
// lanes 16-31: row r,  K = {k0+8..k0+15, k0+24..k0+31}
__device__ __forceinline__ v16h load_frag_g(const _Float16* __restrict__ base,
                                            int row0, int k0, int ldk, int lane) {
    int r  = lane & 15;
    int hi = lane >> 4;
    const _Float16* ptr = base + (size_t)(row0 + r) * ldk + k0 + hi * 8;
    v8h a = *(const v8h*)(ptr);
    v8h b = *(const v8h*)(ptr + 16);
    v16h f;
#pragma unroll
    for (int i = 0; i < 8; ++i) { f[i] = a[i]; f[i + 8] = b[i]; }
    return f;
}

// Fragment from an LDS tile stored row-major, 32 halves per row (64B).
__device__ __forceinline__ v16h load_frag_lds(const _Float16* buf, int row0, int lane) {
    int r  = lane & 15;
    int hi = lane >> 4;
    const _Float16* ptr = buf + (row0 + r) * 32 + hi * 8;
    v8h a = *(const v8h*)(ptr);
    v8h b = *(const v8h*)(ptr + 16);
    v16h f;
#pragma unroll
    for (int i = 0; i < 8; ++i) { f[i] = a[i]; f[i + 8] = b[i]; }
    return f;
}

__device__ __forceinline__ void async_b128(const _Float16* g, const _Float16* l) {
    unsigned lds_off = (unsigned)(unsigned long long)l;   // addr[31:0] = LDS offset
    asm volatile("global_load_async_to_lds_b128 %0, %1, off"
                 :: "v"(lds_off), "v"(g) : "memory");
}

// Stage a 64-row x 32-half B tile into LDS via async DMA (ASYNCcnt).
// 4096 bytes = 256 x b128 chunks; 128 threads -> exactly 2 async ops per
// thread, issued straight-line (no guards -> no exec-mask branches).
__device__ __forceinline__ void stage_B(const _Float16* __restrict__ B, int n0, int k0,
                                        int ldk, _Float16* ldsbuf, int tid) {
    __builtin_assume(tid >= 0 && tid < 128);
    // chunk c = tid: row = c>>2, part = c&3 (16B each)
    {
        int row = tid >> 2, part = tid & 3;
        async_b128(B + (size_t)(n0 + row) * ldk + k0 + part * 8,
                   ldsbuf + row * 32 + part * 8);
    }
    // chunk c = tid + 128: row = 32 + (tid>>2), same part
    {
        int row = 32 + (tid >> 2), part = tid & 3;
        async_b128(B + (size_t)(n0 + row) * ldk + k0 + part * 8,
                   ldsbuf + row * 32 + part * 8);
    }
}

#define WAIT_ASYNC_2() asm volatile("s_wait_asynccnt 2" ::: "memory")
#define WAIT_ASYNC_0() asm volatile("s_wait_asynccnt 0" ::: "memory")

// GEMM1: T7[n, i, jk] = (1/56) * sum_p Xh[n,i,p] * T5[n,jk,p]
__global__ void __launch_bounds__(128)
gemm1_wmma(const _Float16* __restrict__ Xh, const _Float16* __restrict__ T5,
           _Float16* __restrict__ T7) {
    __shared__ _Float16 ldsB[2][64 * 32];
    const int tid  = threadIdx.x;
    const int wave = tid >> 5;
    const int lane = tid & 31;
    const int n    = blockIdx.y;
    const int n0   = blockIdx.x * 64;       // N (jk) tile base, shared by all 4 waves
    const int m0   = wave * 64;             // M (i) tile base, private per wave

    const _Float16* A = Xh + (size_t)n * CC * HW;
    const _Float16* B = T5 + (size_t)n * NK * HW;
    _Float16*       C = T7 + (size_t)n * CC * NK;

    const int NITER = HW / 32;              // 98
    stage_B(B, n0, 0, HW, ldsB[0], tid);

    v8f acc[4][4] = {};
    for (int it = 0; it < NITER; ++it) {
        const int cur = it & 1;
        const int k0  = it * 32;
        if (it + 1 < NITER) {
            stage_B(B, n0, k0 + 32, HW, ldsB[1 - cur], tid);
            WAIT_ASYNC_2();                 // current buffer's 2 ops done (in-order)
        } else {
            WAIT_ASYNC_0();
        }
        __syncthreads();

        v16h af[4], bf[4];
#pragma unroll
        for (int i = 0; i < 4; ++i)  af[i]  = load_frag_g(A, m0 + i * 16, k0, HW, lane);
#pragma unroll
        for (int jt = 0; jt < 4; ++jt) bf[jt] = load_frag_lds(ldsB[cur], jt * 16, lane);
#pragma unroll
        for (int i = 0; i < 4; ++i)
#pragma unroll
            for (int jt = 0; jt < 4; ++jt)
                acc[i][jt] = __builtin_amdgcn_wmma_f32_16x16x32_f16(
                    false, af[i], false, bf[jt], (short)0, acc[i][jt], false, false);
        __syncthreads();                    // buffer 'cur' free for overwrite next+1
    }

    const int col = lane & 15;
    const int hi8 = (lane >> 4) * 8;
#pragma unroll
    for (int i = 0; i < 4; ++i)
#pragma unroll
        for (int jt = 0; jt < 4; ++jt)
#pragma unroll
            for (int v = 0; v < 8; ++v) {
                int row = m0 + i * 16 + hi8 + v;
                C[(size_t)row * NK + n0 + jt * 16 + col] =
                    (_Float16)(acc[i][jt][v] * SCALE1);
            }
}

// GEMM2: out[n, i, p] = (1/sqrt(1792)) * sum_jk T7[n,i,jk] * T6T[n,p,jk]
__global__ void __launch_bounds__(128)
gemm2_wmma(const _Float16* __restrict__ T7, const _Float16* __restrict__ T6T,
           float* __restrict__ out) {
    __shared__ _Float16 ldsB[2][64 * 32];
    const int tid  = threadIdx.x;
    const int wave = tid >> 5;
    const int lane = tid & 31;
    const int n    = blockIdx.y;
    const int n0   = blockIdx.x * 64;       // N (p) tile base
    const int m0   = wave * 64;             // M (i) tile base

    const _Float16* A = T7  + (size_t)n * CC * NK;
    const _Float16* B = T6T + (size_t)n * HW * NK;
    float*          C = out + (size_t)n * CC * HW;

    const int NITER = NK / 32;              // 56
    stage_B(B, n0, 0, NK, ldsB[0], tid);

    v8f acc[4][4] = {};
    for (int it = 0; it < NITER; ++it) {
        const int cur = it & 1;
        const int k0  = it * 32;
        if (it + 1 < NITER) {
            stage_B(B, n0, k0 + 32, NK, ldsB[1 - cur], tid);
            WAIT_ASYNC_2();
        } else {
            WAIT_ASYNC_0();
        }
        __syncthreads();

        v16h af[4], bf[4];
#pragma unroll
        for (int i = 0; i < 4; ++i)  af[i]  = load_frag_g(A, m0 + i * 16, k0, NK, lane);
#pragma unroll
        for (int jt = 0; jt < 4; ++jt) bf[jt] = load_frag_lds(ldsB[cur], jt * 16, lane);
#pragma unroll
        for (int i = 0; i < 4; ++i)
#pragma unroll
            for (int jt = 0; jt < 4; ++jt)
                acc[i][jt] = __builtin_amdgcn_wmma_f32_16x16x32_f16(
                    false, af[i], false, bf[jt], (short)0, acc[i][jt], false, false);
        __syncthreads();
    }

    const int col = lane & 15;
    const int hi8 = (lane >> 4) * 8;
#pragma unroll
    for (int i = 0; i < 4; ++i)
#pragma unroll
        for (int jt = 0; jt < 4; ++jt)
#pragma unroll
            for (int v = 0; v < 8; ++v) {
                int row = m0 + i * 16 + hi8 + v;
                C[(size_t)row * HW + n0 + jt * 16 + col] = acc[i][jt][v] * SCALE2;
            }
}

// ---------------- launch ----------------
extern "C" void kernel_launch(void* const* d_in, const int* in_sizes, int n_in,
                              void* d_out, int out_size, void* d_ws, size_t ws_size,
                              hipStream_t stream) {
    const float* x   = (const float*)d_in[0];   // (8,256,56,56)
    const float* p1w = (const float*)d_in[1];   // (1,256,56,56)
    float*       out = (float*)d_out;           // (8,256,56,56)

    const size_t szXh  = (size_t)NB * CC * HW * sizeof(_Float16);   // 12.8 MB
    const size_t szT5  = (size_t)NB * NK * HW * sizeof(_Float16);   // 89.9 MB
    const size_t szT6T = (size_t)NB * HW * NK * sizeof(_Float16);   // 89.9 MB
    char* ws = (char*)d_ws;
    _Float16* Xh  = (_Float16*)(ws);
    _Float16* T5  = (_Float16*)(ws + szXh);
    _Float16* T6T = (_Float16*)(ws + szXh + szT5);
    _Float16* T7  = (_Float16*)(ws + szXh + szT5 + szT6T);
    (void)szT6T; (void)ws_size; (void)in_sizes; (void)n_in; (void)out_size;

    const long long nXh = (long long)NB * CC * HW;       // 6.4M
    const long long nT5 = (long long)NB * NK * HW;       // 45.2M
    prep_xh <<<(unsigned)((nXh + 255) / 256), 256, 0, stream>>>(x, Xh);
    prep_t5 <<<(unsigned)((nT5 + 255) / 256), 256, 0, stream>>>(x, p1w, T5);
    prep_t6t<<<(unsigned)((nT5 + 255) / 256), 256, 0, stream>>>(x, p1w, T6T);

    dim3 g1(NK / 64, NB);   // 28 x 8
    gemm1_wmma<<<g1, 128, 0, stream>>>(Xh, T5, T7);
    dim3 g2(HW / 64, NB);   // 49 x 8
    gemm2_wmma<<<g2, 128, 0, stream>>>(T7, T6T, out);
}
